// Attention_47768626266365
// MI455X (gfx1250) — matrix operand order repaired and checked by reading, not compiled, not run on hardware
//
#include <hip/hip_runtime.h>

#define BATCH 16
#define CDIM 512
#define NPIX 1024          // 32*32
#define HEADS 8
#define DH 64
#define INNER 512          // HEADS*DH
#define QKVO 1536          // 3*INNER

typedef __attribute__((ext_vector_type(16))) __bf16 v16bf;
typedef __attribute__((ext_vector_type(8)))  float  v8f;
typedef __attribute__((ext_vector_type(4))) unsigned tdm_u4;
typedef __attribute__((ext_vector_type(8))) int      tdm_i8;
typedef __attribute__((ext_vector_type(4))) int      tdm_i4;

struct BF16x16 { union { v16bf v; uint4 q[2]; unsigned short s[16]; }; };

__device__ __forceinline__ unsigned short f2bf(float f) {
    unsigned u = __float_as_uint(f);
    u += 0x7FFFu + ((u >> 16) & 1u);          // round-to-nearest-even
    return (unsigned short)(u >> 16);
}
__device__ __forceinline__ float bf2f(unsigned short h) {
    return __uint_as_float(((unsigned)h) << 16);
}
// load the 16-bit A-matrix lane pattern: two contiguous groups of 8 bf16
__device__ __forceinline__ v16bf load2x8(const unsigned short* p0, const unsigned short* p1) {
    BF16x16 t;
    t.q[0] = *(const uint4*)p0;
    t.q[1] = *(const uint4*)p1;
    return t.v;
}
// load a B-matrix lane: 16 contiguous bf16 (one K half-column, this code's k = 16h+s order)
__device__ __forceinline__ v16bf load16(const unsigned short* p) {
    BF16x16 t;
    t.q[0] = *(const uint4*)p;
    t.q[1] = *(const uint4*)(p + 8);
    return t.v;
}
#define WMMA_BF16(A, B, C) \
    __builtin_amdgcn_wmma_f32_16x16x32_bf16(false, (A), false, (B), (short)0, (C), false, false)

// ---------------------------------------------------------------------------
// Tensor Data Mover: DMA a 2-D bf16 tile (tile_d0 elems x tile_d1 rows) from
// global (row stride stride_d0 elems) into LDS, contiguous row-major.
// D# layout per CDNA5 ISA ch.8: group0 {count=1, lds_addr, global_addr, type=2},
// group1 {data_size=2B, tensor dims, tile dims, dim0 stride}.
__device__ __forceinline__ void tdm_load_tile_bf16(
        unsigned lds_byte_addr, const unsigned short* gptr,
        unsigned tile_d0, unsigned tile_d1,
        unsigned tensor_d0, unsigned tensor_d1, unsigned stride_d0) {
    unsigned long long ga = (unsigned long long)gptr;
    tdm_u4 g0;
    g0[0] = 1u;                                             // count=1, user D#
    g0[1] = lds_byte_addr;                                  // lds_addr
    g0[2] = (unsigned)(ga & 0xFFFFFFFFu);                   // global_addr lo
    g0[3] = (unsigned)((ga >> 32) & 0x01FFFFFFu) | (2u << 30); // addr hi | type=2
    tdm_i8 g1;
    g1[0] = (int)(1u << 16);                                // data_size=1 (2 bytes)
    g1[1] = (int)((tensor_d0 & 0xFFFFu) << 16);             // tensor_dim0[15:0]
    g1[2] = (int)(((tensor_d0 >> 16) & 0xFFFFu) | ((tensor_d1 & 0xFFFFu) << 16));
    g1[3] = (int)(((tensor_d1 >> 16) & 0xFFFFu) | ((tile_d0 & 0xFFFFu) << 16));
    g1[4] = (int)(tile_d1 & 0xFFFFu);                       // tile_dim1, tile_dim2=0
    g1[5] = (int)stride_d0;                                 // dim0_stride[31:0]
    g1[6] = 0;
    g1[7] = 0;
    tdm_i4 gz = {0, 0, 0, 0};
#if defined(__clang_major__) && (__clang_major__ >= 23)
    tdm_i8 gz8 = {0, 0, 0, 0, 0, 0, 0, 0};
    __builtin_amdgcn_tensor_load_to_lds(g0, g1, gz, gz, gz8, 0);
#else
    __builtin_amdgcn_tensor_load_to_lds(g0, g1, gz, gz, 0);
#endif
}
__device__ __forceinline__ unsigned lds_addr_of(const void* p) {
    return (unsigned)(unsigned long long)p;   // low 32 bits of flat addr = LDS byte addr
}

// ---------------------------------------------------------------------------
// 0) cast weights f32 -> bf16
__global__ __launch_bounds__(256) void cvt_weights(
        const float* __restrict__ wqkv, const float* __restrict__ wout,
        unsigned short* __restrict__ wqkv_bf, unsigned short* __restrict__ wout_bf) {
    int i = blockIdx.x * 256 + threadIdx.x;
    if (i < QKVO * CDIM) wqkv_bf[i] = f2bf(wqkv[i]);
    int j = i - QKVO * CDIM;
    if (j >= 0 && j < CDIM * INNER) wout_bf[j] = f2bf(wout[j]);
}

// ---------------------------------------------------------------------------
// 1) channel layernorm: x[b][c][p] f32 -> xn[b][p][c] bf16 (pixel-major)
__global__ __launch_bounds__(256) void chan_ln(
        const float* __restrict__ x, const float* __restrict__ gamma,
        unsigned short* __restrict__ Xn) {
    int wave = threadIdx.x >> 5, lane = threadIdx.x & 31;
    int pix = blockIdx.x * 8 + wave;          // global pixel over B*N
    int b = pix >> 10, p = pix & 1023;
    const float* xb = x + ((size_t)b * CDIM) * NPIX + p;
    float vals[16]; float s = 0.f, s2 = 0.f;
#pragma unroll
    for (int i = 0; i < 16; ++i) {
        float t = xb[(size_t)(lane + i * 32) * NPIX];
        vals[i] = t; s += t; s2 += t * t;
    }
#pragma unroll
    for (int m = 1; m < 32; m <<= 1) { s += __shfl_xor(s, m, 32); s2 += __shfl_xor(s2, m, 32); }
    float mean = s * (1.f / 512.f);
    float var  = s2 * (1.f / 512.f) - mean * mean;
    float r = rsqrtf(var + 1e-5f);
    unsigned short* xo = Xn + ((size_t)b * NPIX + p) * CDIM;
#pragma unroll
    for (int i = 0; i < 16; ++i) {
        int c = lane + i * 32;
        xo[c] = f2bf((vals[i] - mean) * r * gamma[c]);
    }
}

// ---------------------------------------------------------------------------
// 2) QKV GEMM with TDM-staged, double-buffered weight tiles in LDS.
//    qkv[b][o][p] = sum_c xn[b][p][c] * Wqkv[o][c]
__global__ __launch_bounds__(256) void qkv_gemm(
        const unsigned short* __restrict__ Xn,     // [B][N][C] bf16
        const unsigned short* __restrict__ Wq,     // [1536][512] bf16
        unsigned short* __restrict__ QKV) {        // [B][1536][N] bf16
    __shared__ __align__(16) unsigned short sW[2][128][32];  // 16 KB double buffer
    const int wave = threadIdx.x >> 5, lane = threadIdx.x & 31;
    const int ln = lane & 15, hi = lane >> 4;
    const int b = blockIdx.z;
    const int m0 = blockIdx.x * 64 + (wave & 3) * 16;        // pixel rows
    const int nblk = blockIdx.y * 128;                       // block's out-channel base
    const int nw = (wave >> 2) * 64;                         // wave's 64-wide N slice
    const unsigned short* Xb = Xn + ((size_t)b * NPIX + m0 + ln) * CDIM;
    const bool issuer = (threadIdx.x < 32);                  // wave-uniform

    if (issuer)
        tdm_load_tile_bf16(lds_addr_of(&sW[0][0][0]), Wq + (size_t)nblk * CDIM,
                           32, 128, CDIM, QKVO, CDIM);
    v8f acc[4] = {};
    for (int kc = 0; kc < CDIM; kc += 32) {
        const int ib = (kc >> 5) & 1;
        if (issuer) __builtin_amdgcn_s_wait_tensorcnt(0);    // buffer ib DMA done
        __syncthreads();                                     // publish ib, retire ib^1
        if (issuer && (kc + 32 < CDIM))
            tdm_load_tile_bf16(lds_addr_of(&sW[ib ^ 1][0][0]),
                               Wq + (size_t)nblk * CDIM + kc + 32,
                               32, 128, CDIM, QKVO, CDIM);
        v16bf a = load2x8(Xb + kc + hi * 16, Xb + kc + hi * 16 + 8);
#pragma unroll
        for (int t = 0; t < 4; ++t)
            acc[t] = WMMA_BF16(a, load16(&sW[ib][nw + t * 16 + ln][hi * 16]), acc[t]);
    }
#pragma unroll
    for (int t = 0; t < 4; ++t) {
        int o = nblk + nw + t * 16 + ln;
        unsigned short* qp = QKV + ((size_t)b * QKVO + o) * NPIX + m0 + 8 * hi;
        BF16x16 st;
#pragma unroll
        for (int v = 0; v < 8; ++v) st.s[v] = f2bf(acc[t][v]);
        *(uint4*)qp = st.q[0];
    }
}

// ---------------------------------------------------------------------------
// 3) depthwise 3x3 + bias, reshape into head layouts; scale folded into Q
__global__ __launch_bounds__(256) void dwconv_kernel(
        const unsigned short* __restrict__ QKV,    // [B][1536][1024] bf16
        const float* __restrict__ wq, const float* __restrict__ bq,
        const float* __restrict__ wk, const float* __restrict__ bk,
        const float* __restrict__ wv, const float* __restrict__ bv,
        unsigned short* __restrict__ Qo,           // [B][H][N][D]
        unsigned short* __restrict__ Ko,           // [B][H][N][D]
        unsigned short* __restrict__ Vo) {         // [B][H][D][N]
    size_t i = (size_t)blockIdx.x * 256 + threadIdx.x;   // < B*1536*1024
    int p = (int)(i & 1023);
    int och = (int)((i >> 10) % QKVO);
    int b = (int)(i / ((size_t)QKVO * NPIX));
    int sec = och >> 9, ch = och & 511;
    int X = p >> 5, Y = p & 31;
    const float* w; const float* bias;
    if (sec == 0)      { w = wq; bias = bq; }
    else if (sec == 1) { w = wk; bias = bk; }
    else               { w = wv; bias = bv; }
    const unsigned short* in = QKV + ((size_t)b * QKVO + och) * NPIX;
    float acc = bias[ch];
#pragma unroll
    for (int di = 0; di < 3; ++di) {
        int xx = X + di - 1;
        if ((unsigned)xx >= 32u) continue;
#pragma unroll
        for (int dj = 0; dj < 3; ++dj) {
            int yy = Y + dj - 1;
            if ((unsigned)yy >= 32u) continue;
            acc += w[ch * 9 + di * 3 + dj] * bf2f(in[xx * 32 + yy]);
        }
    }
    int h = ch >> 6, d = ch & 63;
    if (sec == 0)
        Qo[(((size_t)b * HEADS + h) * NPIX + p) * DH + d] = f2bf(acc * 0.125f);
    else if (sec == 1)
        Ko[(((size_t)b * HEADS + h) * NPIX + p) * DH + d] = f2bf(acc);
    else
        Vo[(((size_t)b * HEADS + h) * DH + d) * NPIX + p] = f2bf(acc);
}

// ---------------------------------------------------------------------------
// 4) flash attention per (b, h, 128-query tile); one wave = 16 query rows
__global__ __launch_bounds__(256) void attn_kernel(
        const unsigned short* __restrict__ Q,      // [B][H][N][D] (pre-scaled)
        const unsigned short* __restrict__ K,      // [B][H][N][D]
        const unsigned short* __restrict__ V,      // [B][H][D][N]
        const float* __restrict__ pos_emb,         // [3969][8]
        unsigned short* __restrict__ O) {          // [B][N][H*D] bf16
    __shared__ __align__(16) unsigned short sP[8][16][32];   // per-wave P transpose buffer
    const int wave = threadIdx.x >> 5, lane = threadIdx.x & 31;
    const int ln = lane & 15, hi = lane >> 4;
    const int b = blockIdx.z, h = blockIdx.y;
    const int q0 = blockIdx.x * 128 + wave * 16;

    const unsigned short* Qb = Q + (((size_t)b * HEADS + h) * NPIX) * DH;
    const unsigned short* Kb = K + (((size_t)b * HEADS + h) * NPIX) * DH;
    const unsigned short* Vb = V + (((size_t)b * HEADS + h) * DH) * NPIX;

    const unsigned short* qrow = Qb + (size_t)(q0 + ln) * DH;
    v16bf aq0 = load2x8(qrow + hi * 16,      qrow + 8 + hi * 16);
    v16bf aq1 = load2x8(qrow + 32 + hi * 16, qrow + 40 + hi * 16);

    v8f oa0 = {}, oa1 = {}, oa2 = {}, oa3 = {};
    float mrow[8], lrow[8];
#pragma unroll
    for (int v = 0; v < 8; ++v) { mrow[v] = -3.0e38f; lrow[v] = 0.f; }

    for (int j0 = 0; j0 < NPIX; j0 += 32) {
        if (j0 + 32 < NPIX) {   // warm caches for the next tile
            __builtin_prefetch(Kb + (size_t)(j0 + 32 + ln) * DH, 0, 1);
            __builtin_prefetch(Vb + (size_t)ln * NPIX + j0 + 32, 0, 1);
        }
        // S = Q*K^T, two 16-wide column tiles, K-dim = 64 (2 wmma each)
        const unsigned short* kp0 = Kb + (size_t)(j0 + ln) * DH + hi * 16;
        const unsigned short* kp1 = Kb + (size_t)(j0 + 16 + ln) * DH + hi * 16;
        v8f s0 = {}, s1 = {};
        s0 = WMMA_BF16(aq0, load16(kp0),      s0);
        s0 = WMMA_BF16(aq1, load16(kp0 + 32), s0);
        s1 = WMMA_BF16(aq0, load16(kp1),      s1);
        s1 = WMMA_BF16(aq1, load16(kp1 + 32), s1);

        // bias + row-wise online softmax (rows live in VGPR index v)
        float pmax[8];
#pragma unroll
        for (int v = 0; v < 8; ++v) {
            int i  = q0 + v + 8 * hi;
            int j  = j0 + ln;
            int jB = j0 + 16 + ln;
            int idxA = ((i >> 5) - (j  >> 5) + 31) * 63 + ((i & 31) - (j  & 31) + 31);
            int idxB = ((i >> 5) - (jB >> 5) + 31) * 63 + ((i & 31) - (jB & 31) + 31);
            s0[v] += pos_emb[idxA * 8 + h];
            s1[v] += pos_emb[idxB * 8 + h];
            pmax[v] = fmaxf(s0[v], s1[v]);
        }
#pragma unroll
        for (int v = 0; v < 8; ++v) {
#pragma unroll
            for (int m = 1; m < 16; m <<= 1) pmax[v] = fmaxf(pmax[v], __shfl_xor(pmax[v], m, 32));
        }
        float scale[8];
#pragma unroll
        for (int v = 0; v < 8; ++v) {
            float mn = fmaxf(mrow[v], pmax[v]);
            scale[v] = __expf(mrow[v] - mn);
            mrow[v] = mn;
            float p0 = __expf(s0[v] - mn);
            float p1 = __expf(s1[v] - mn);
            s0[v] = p0; s1[v] = p1;
            float rs = p0 + p1;
#pragma unroll
            for (int m = 1; m < 16; m <<= 1) rs += __shfl_xor(rs, m, 32);
            lrow[v] = lrow[v] * scale[v] + rs;
        }
#pragma unroll
        for (int v = 0; v < 8; ++v) {
            oa0[v] *= scale[v]; oa1[v] *= scale[v];
            oa2[v] *= scale[v]; oa3[v] *= scale[v];
        }
        // transpose P (C/D layout -> A layout) through per-wave LDS tile
#pragma unroll
        for (int v = 0; v < 8; ++v) {
            sP[wave][v + 8 * hi][ln]      = f2bf(s0[v]);
            sP[wave][v + 8 * hi][16 + ln] = f2bf(s1[v]);
        }
        __asm volatile("" ::: "memory");   // DS ops are in-order per wave
        v16bf pa = load2x8(&sP[wave][ln][hi * 16], &sP[wave][ln][hi * 16 + 8]);
        __asm volatile("" ::: "memory");

        // O += P * V  (K-dim = 32 keys, four 16-wide d tiles)
        const unsigned short* vp = Vb + (size_t)ln * NPIX + j0 + hi * 16;
        oa0 = WMMA_BF16(pa, load16(vp),               oa0);
        oa1 = WMMA_BF16(pa, load16(vp + 16 * NPIX),   oa1);
        oa2 = WMMA_BF16(pa, load16(vp + 32 * NPIX),   oa2);
        oa3 = WMMA_BF16(pa, load16(vp + 48 * NPIX),   oa3);
    }
#pragma unroll
    for (int v = 0; v < 8; ++v) {
        float inv = 1.0f / lrow[v];
        oa0[v] *= inv; oa1[v] *= inv; oa2[v] *= inv; oa3[v] *= inv;
    }
    unsigned short* Ob = O + ((size_t)b * NPIX) * INNER + h * DH;
#pragma unroll
    for (int v = 0; v < 8; ++v) {
        int p = q0 + v + 8 * hi;
        unsigned short* op = Ob + (size_t)p * INNER;
        op[ln]      = f2bf(oa0[v]);
        op[16 + ln] = f2bf(oa1[v]);
        op[32 + ln] = f2bf(oa2[v]);
        op[48 + ln] = f2bf(oa3[v]);
    }
}

// ---------------------------------------------------------------------------
// 5) output projection with TDM-staged weights: out[b][o][p] = sum_c AO[b][p][c]*Wout[o][c]
__global__ __launch_bounds__(256) void outproj_gemm(
        const unsigned short* __restrict__ AO,     // [B][N][512] bf16
        const unsigned short* __restrict__ Wo,     // [512][512] bf16
        float* __restrict__ Out) {                 // [B][512][N] f32
    __shared__ __align__(16) unsigned short sW[2][128][32];
    const int wave = threadIdx.x >> 5, lane = threadIdx.x & 31;
    const int ln = lane & 15, hi = lane >> 4;
    const int b = blockIdx.z;
    const int m0 = blockIdx.x * 64 + (wave & 3) * 16;
    const int nblk = blockIdx.y * 128;
    const int nw = (wave >> 2) * 64;
    const unsigned short* Ab = AO + ((size_t)b * NPIX + m0 + ln) * INNER;
    const bool issuer = (threadIdx.x < 32);

    if (issuer)
        tdm_load_tile_bf16(lds_addr_of(&sW[0][0][0]), Wo + (size_t)nblk * INNER,
                           32, 128, INNER, INNER, INNER);
    v8f acc[4] = {};
    for (int kc = 0; kc < INNER; kc += 32) {
        const int ib = (kc >> 5) & 1;
        if (issuer) __builtin_amdgcn_s_wait_tensorcnt(0);
        __syncthreads();
        if (issuer && (kc + 32 < INNER))
            tdm_load_tile_bf16(lds_addr_of(&sW[ib ^ 1][0][0]),
                               Wo + (size_t)nblk * INNER + kc + 32,
                               32, 128, INNER, INNER, INNER);
        v16bf a = load2x8(Ab + kc + hi * 16, Ab + kc + hi * 16 + 8);
#pragma unroll
        for (int t = 0; t < 4; ++t)
            acc[t] = WMMA_BF16(a, load16(&sW[ib][nw + t * 16 + ln][hi * 16]), acc[t]);
    }
#pragma unroll
    for (int t = 0; t < 4; ++t) {
        int o = nblk + nw + t * 16 + ln;
        float* op = Out + ((size_t)b * CDIM + o) * NPIX + m0 + 8 * hi;
        float4 f0 = make_float4(acc[t][0], acc[t][1], acc[t][2], acc[t][3]);
        float4 f1 = make_float4(acc[t][4], acc[t][5], acc[t][6], acc[t][7]);
        *(float4*)op = f0;
        *(float4*)(op + 4) = f1;
    }
}

// ---------------------------------------------------------------------------
// workspace layout (bytes)
#define OFF_XN    ((size_t)0)                       // 16 MB  xn bf16 [B][N][C]
#define OFF_WQKV  ((size_t)16777216)                // 1.5 MB
#define OFF_WOUT  ((size_t)18350080)                // 0.5 MB
#define OFF_QKVB  ((size_t)18874368)                // 48 MB  qkv bf16 [B][1536][N]
#define OFF_QS    ((size_t)69206016)                // 16 MB
#define OFF_KS    ((size_t)85983232)                // 16 MB
#define OFF_VS    ((size_t)102760448)               // 16 MB
#define OFF_AO    ((size_t)119537664)               // 16 MB

extern "C" void kernel_launch(void* const* d_in, const int* in_sizes, int n_in,
                              void* d_out, int out_size, void* d_ws, size_t ws_size,
                              hipStream_t stream) {
    const float* x       = (const float*)d_in[0];
    const float* gamma   = (const float*)d_in[1];
    const float* w_qkv   = (const float*)d_in[2];
    const float* dw_w_q  = (const float*)d_in[3];
    const float* dw_b_q  = (const float*)d_in[4];
    const float* dw_w_k  = (const float*)d_in[5];
    const float* dw_b_k  = (const float*)d_in[6];
    const float* dw_w_v  = (const float*)d_in[7];
    const float* dw_b_v  = (const float*)d_in[8];
    const float* w_out   = (const float*)d_in[9];
    const float* pos_emb = (const float*)d_in[10];
    float* out = (float*)d_out;

    char* ws = (char*)d_ws;
    unsigned short* xn      = (unsigned short*)(ws + OFF_XN);
    unsigned short* wqkv_bf = (unsigned short*)(ws + OFF_WQKV);
    unsigned short* wout_bf = (unsigned short*)(ws + OFF_WOUT);
    unsigned short* qkvb    = (unsigned short*)(ws + OFF_QKVB);
    unsigned short* qs      = (unsigned short*)(ws + OFF_QS);
    unsigned short* ks      = (unsigned short*)(ws + OFF_KS);
    unsigned short* vs      = (unsigned short*)(ws + OFF_VS);
    unsigned short* ao      = (unsigned short*)(ws + OFF_AO);

    cvt_weights<<<4096, 256, 0, stream>>>(w_qkv, w_out, wqkv_bf, wout_bf);
    chan_ln<<<(BATCH * NPIX) / 8, 256, 0, stream>>>(x, gamma, xn);
    qkv_gemm<<<dim3(NPIX / 64, QKVO / 128, BATCH), 256, 0, stream>>>(xn, wqkv_bf, qkvb);
    dwconv_kernel<<<(BATCH * QKVO * NPIX) / 256, 256, 0, stream>>>(
        qkvb, dw_w_q, dw_b_q, dw_w_k, dw_b_k, dw_w_v, dw_b_v, qs, ks, vs);
    attn_kernel<<<dim3(NPIX / 128, HEADS, BATCH), 256, 0, stream>>>(qs, ks, vs, pos_emb, ao);
    outproj_gemm<<<dim3(NPIX / 64, INNER / 128, BATCH), 256, 0, stream>>>(ao, wout_bf, out);
}
